// Linear_63342177681727
// MI455X (gfx1250) — matrix operand order repaired and checked by reading, not compiled, not run on hardware
//
#include <hip/hip_runtime.h>

// ---------------------------------------------------------------------------
// LoRA linear: out = x @ W^T + b + (scaling * dropout(x)) @ A[aid]^T @ Bw[aid]^T
// M = 16384 (B*S), N = 4096 (DOUT), K = 4096 (DIN), R = 16, NA = 4
//
// Pipeline (fast path, needs ~162 MB workspace):
//   cvt_bf16_kernel x2 : x, W  ->  bf16 copies in workspace (one-time)
//   lora_h_kernel      : h[M,16] = (dropout(x)*2/0.95) @ A[aid]^T   (bf16 WMMA)
//   lora_gemm_tdm      : 128x128 tile GEMM; K-loop staging via TDM
//                        tensor_load_to_lds (double-buffered, wave0 issues,
//                        s_wait_tensorcnt + barrier), bf16 WMMA, epilogue adds
//                        bias + one zero-padded WMMA for h @ Bw^T.
// Fallback (small ws): lora_gemm_kernel converts f32->bf16 in-kernel.
// ---------------------------------------------------------------------------

typedef __bf16  v16bf __attribute__((ext_vector_type(16)));
typedef float   v8f   __attribute__((ext_vector_type(8)));
typedef unsigned int u32x4 __attribute__((ext_vector_type(4)));
typedef int          i32x4 __attribute__((ext_vector_type(4)));
typedef int          i32x8 __attribute__((ext_vector_type(8)));

#define DIN   4096
#define DOUT  4096
#define MROWS 16384
#define P_DROP 0.05f
#define KEEP_SCALE (2.0f / 0.95f)   // (1/(1-p)) * (alpha/r)

union Frag {
    v16bf    v;
    unsigned u[8];
};

// round-to-nearest-even f32 -> bf16, packed pair
__device__ __forceinline__ unsigned bf2pack(float lo, float hi) {
    union { float f; unsigned u; } a, b;
    a.f = lo; b.f = hi;
    unsigned ra = (a.u + 0x7FFFu + ((a.u >> 16) & 1u)) >> 16;
    unsigned rb = (b.u + 0x7FFFu + ((b.u >> 16) & 1u)) >> 16;
    return (ra & 0xFFFFu) | (rb << 16);
}

// ---------------------------------------------------------------------------
// Bulk f32 -> bf16 conversion (8 elements / thread)
// ---------------------------------------------------------------------------
__global__ __launch_bounds__(256) void cvt_bf16_kernel(
    const float* __restrict__ src, unsigned short* __restrict__ dst, int n8)
{
    int idx = blockIdx.x * 256 + threadIdx.x;
    if (idx >= n8) return;
    const float4* s = (const float4*)src + (size_t)idx * 2;
    float4 a = s[0], b = s[1];
    ((uint4*)dst)[idx] = make_uint4(bf2pack(a.x, a.y), bf2pack(a.z, a.w),
                                    bf2pack(b.x, b.y), bf2pack(b.z, b.w));
}

// ---------------------------------------------------------------------------
// TDM: load one 128-row x 32-col bf16 tile (row stride 4096 elems in memory)
// into LDS with 16B padding after each 64B row (-> 80B LDS row stride).
// D# layout per CDNA5 ISA 8.3-8.5.
// ---------------------------------------------------------------------------
__device__ __forceinline__ void tdm_load_tile_bf16(
    unsigned lds_byte_addr, const unsigned short* gptr, unsigned tensor_rows)
{
    unsigned long long ga = (unsigned long long)(uintptr_t)gptr;
    u32x4 g0;
    g0[0] = 1u;                                          // count=1, user mode
    g0[1] = lds_byte_addr;                               // lds_addr
    g0[2] = (unsigned)ga;                                // global_addr[31:0]
    g0[3] = (unsigned)((ga >> 32) & 0x01FFFFFFu)         // global_addr[56:32]
          | (2u << 30);                                  // type=2 ("image")
    i32x8 g1;
    g1[0] = (int)((1u << 16)      // data_size = 1 -> 2 bytes
                | (1u << 20)      // pad_enable
                | (3u << 22)      // pad_interval code 3 -> every 16 DWORDs (64B row)
                | (3u << 25));    // pad_amount  code 3 -> 4 DWORDs (16B)
    g1[1] = (int)(4096u << 16);                          // tensor_dim0 = 4096 (lo16)
    g1[2] = (int)((tensor_rows & 0xFFFFu) << 16);        // dim0 hi=0 | tensor_dim1 lo
    g1[3] = (int)((tensor_rows >> 16) | (32u << 16));    // tensor_dim1 hi | tile_dim0=32
    g1[4] = (int)128u;                                   // tile_dim1=128, tile_dim2=0
    g1[5] = (int)4096u;                                  // tensor_dim0_stride lo
    g1[6] = 0;                                           // stride0 hi | stride1 lo
    g1[7] = 0;
    i32x4 z4 = {0, 0, 0, 0};
#if __clang_major__ >= 23
    i32x8 z8 = {0, 0, 0, 0, 0, 0, 0, 0};
    __builtin_amdgcn_tensor_load_to_lds(g0, g1, z4, z4, z8, 0);
#else
    __builtin_amdgcn_tensor_load_to_lds(g0, g1, z4, z4, 0);
#endif
}

// ---------------------------------------------------------------------------
// Kernel A: h[row, r] = sum_k xd[row,k] * A[aid, r, k]
// One wave per 16-row group: M=16 x N=16(r) x K=4096, WMMA bf16 per 32-K step.
// ---------------------------------------------------------------------------
__global__ __launch_bounds__(256) void lora_h_kernel(
    const float* __restrict__ x,
    const float* __restrict__ Alora,     // [NA,16,DIN]
    const float* __restrict__ drop_u,
    const int*   __restrict__ adapter_id,
    float*       __restrict__ h)         // [M,16]
{
    const int lane = threadIdx.x & 31;
    const int wave = threadIdx.x >> 5;
    const int group = blockIdx.x * 8 + wave;          // 0..1023
    const int row0  = group * 16;
    const int aid   = adapter_id[row0 >> 11];         // 2048 rows per batch
    const int tm    = lane & 15;
    const int sel   = lane >> 4;

    const float* xrow = x      + (size_t)(row0 + tm) * DIN;
    const float* urow = drop_u + (size_t)(row0 + tm) * DIN;
    const float* arow = Alora  + ((size_t)aid * 16 + tm) * DIN;

    v8f acc = {};

    for (int k0 = 0; k0 < DIN; k0 += 32) {
        Frag fa, fb;
        #pragma unroll
        for (int c = 0; c < 2; ++c) {
            const int kk = k0 + sel * 16 + c * 8;
            float4 p0 = *(const float4*)(xrow + kk);
            float4 p1 = *(const float4*)(xrow + kk + 4);
            float4 q0 = *(const float4*)(urow + kk);
            float4 q1 = *(const float4*)(urow + kk + 4);
            float e0 = p0.x * ((q0.x >= P_DROP) ? KEEP_SCALE : 0.f);
            float e1 = p0.y * ((q0.y >= P_DROP) ? KEEP_SCALE : 0.f);
            float e2 = p0.z * ((q0.z >= P_DROP) ? KEEP_SCALE : 0.f);
            float e3 = p0.w * ((q0.w >= P_DROP) ? KEEP_SCALE : 0.f);
            float e4 = p1.x * ((q1.x >= P_DROP) ? KEEP_SCALE : 0.f);
            float e5 = p1.y * ((q1.y >= P_DROP) ? KEEP_SCALE : 0.f);
            float e6 = p1.z * ((q1.z >= P_DROP) ? KEEP_SCALE : 0.f);
            float e7 = p1.w * ((q1.w >= P_DROP) ? KEEP_SCALE : 0.f);
            fa.u[c * 4 + 0] = bf2pack(e0, e1);
            fa.u[c * 4 + 1] = bf2pack(e2, e3);
            fa.u[c * 4 + 2] = bf2pack(e4, e5);
            fa.u[c * 4 + 3] = bf2pack(e6, e7);
        }
        {
            const int kk = k0 + sel * 16;
            float4 a0 = *(const float4*)(arow + kk);
            float4 a1 = *(const float4*)(arow + kk + 4);
            float4 a2 = *(const float4*)(arow + kk + 8);
            float4 a3 = *(const float4*)(arow + kk + 12);
            fb.u[0] = bf2pack(a0.x, a0.y); fb.u[1] = bf2pack(a0.z, a0.w);
            fb.u[2] = bf2pack(a1.x, a1.y); fb.u[3] = bf2pack(a1.z, a1.w);
            fb.u[4] = bf2pack(a2.x, a2.y); fb.u[5] = bf2pack(a2.z, a2.w);
            fb.u[6] = bf2pack(a3.x, a3.y); fb.u[7] = bf2pack(a3.z, a3.w);
        }
        acc = __builtin_amdgcn_wmma_f32_16x16x32_bf16(
            false, fa.v, false, fb.v, (short)0, acc, false, false);
    }

    const int mb = sel * 8;
    #pragma unroll
    for (int v = 0; v < 8; ++v)
        h[(size_t)(row0 + mb + v) * 16 + tm] = acc[v];
}

// ---------------------------------------------------------------------------
// Shared epilogue + fragment helpers for the 128x128 GEMM kernels
// ---------------------------------------------------------------------------
#define BM 128
#define BN 128
#define BK 32
#define LDA 40   // LDS row stride in bf16 elements (80B)

// ---------------------------------------------------------------------------
// Kernel B (fast path): bf16 operands in global, TDM staging, WMMA compute
// ---------------------------------------------------------------------------
__global__ __launch_bounds__(256) void lora_gemm_tdm_kernel(
    const unsigned short* __restrict__ xb,   // [M, DIN]  bf16
    const unsigned short* __restrict__ wb,   // [DOUT, DIN] bf16
    const float* __restrict__ bias,
    const float* __restrict__ Bw,            // [NA, DOUT, 16] f32
    const int*   __restrict__ adapter_id,
    const float* __restrict__ h,             // [M, 16] f32
    float*       __restrict__ out)
{
    __shared__ unsigned short a_sm[2][BM * LDA];
    __shared__ unsigned short b_sm[2][BN * LDA];

    const int t    = threadIdx.x;
    const int lane = t & 31;
    const int wave = t >> 5;
    const int tm   = lane & 15;
    const int sel  = lane >> 4;
    const int wm   = wave & 3;
    const int wn   = wave >> 2;

    const int m0 = blockIdx.y * BM;
    const int n0 = blockIdx.x * BN;

    const unsigned ldsA0 = (unsigned)(uintptr_t)&a_sm[0][0];
    const unsigned ldsA1 = (unsigned)(uintptr_t)&a_sm[1][0];
    const unsigned ldsB0 = (unsigned)(uintptr_t)&b_sm[0][0];
    const unsigned ldsB1 = (unsigned)(uintptr_t)&b_sm[1][0];

    const unsigned short* xg = xb + (size_t)m0 * DIN;
    const unsigned short* wg = wb + (size_t)n0 * DIN;

    v8f zf = {};
    v8f acc[2][4];
    #pragma unroll
    for (int i = 0; i < 2; ++i)
        #pragma unroll
        for (int j = 0; j < 4; ++j) acc[i][j] = zf;

    // prologue: wave0 kicks off first tile pair
    if (wave == 0) {
        tdm_load_tile_bf16(ldsA0, xg, MROWS);
        tdm_load_tile_bf16(ldsB0, wg, DOUT);
    }

    int cur = 0;
    for (int k0 = 0; k0 < DIN; k0 += BK, cur ^= 1) {
        if (wave == 0)
            __builtin_amdgcn_s_wait_tensorcnt(0);   // tile `cur` landed in LDS
        __syncthreads();                            // publish to all waves
        if (wave == 0 && (k0 + BK) < DIN) {
            // overlap: start DMA of next tile into the other buffer
            tdm_load_tile_bf16(cur ? ldsA0 : ldsA1, xg + k0 + BK, MROWS);
            tdm_load_tile_bf16(cur ? ldsB0 : ldsB1, wg + k0 + BK, DOUT);
        }

        const unsigned short* abuf = a_sm[cur];
        const unsigned short* bbuf = b_sm[cur];
        Frag fa[2], fb[4];
        #pragma unroll
        for (int i = 0; i < 2; ++i) {
            const unsigned short* base = &abuf[(wm * 32 + i * 16 + tm) * LDA];
            *(uint4*)&fa[i].u[0] = *(const uint4*)(base + sel * 8);
            *(uint4*)&fa[i].u[4] = *(const uint4*)(base + 16 + sel * 8);
        }
        #pragma unroll
        for (int j = 0; j < 4; ++j) {
            const unsigned short* base = &bbuf[(wn * 64 + j * 16 + tm) * LDA];
            *(uint4*)&fb[j].u[0] = *(const uint4*)(base + sel * 8);
            *(uint4*)&fb[j].u[4] = *(const uint4*)(base + 16 + sel * 8);
        }
        #pragma unroll
        for (int i = 0; i < 2; ++i)
            #pragma unroll
            for (int j = 0; j < 4; ++j)
                acc[i][j] = __builtin_amdgcn_wmma_f32_16x16x32_bf16(
                    false, fa[i].v, false, fb[j].v, (short)0, acc[i][j],
                    false, false);
    }

    // ---------------- LoRA epilogue: acc += h_tile @ Bw_tile^T ----------------
    const int aid   = adapter_id[m0 >> 11];
    const int lrow  = t >> 1;
    const int lhalf = t & 1;
    __syncthreads();
    if (lhalf == 0) {
        const float* hp = h + (size_t)(m0 + lrow) * 16;
        float4 h0 = *(const float4*)(hp);
        float4 h1 = *(const float4*)(hp + 4);
        float4 h2 = *(const float4*)(hp + 8);
        float4 h3 = *(const float4*)(hp + 12);
        unsigned* ap = (unsigned*)&a_sm[0][lrow * LDA];
        *(uint4*)(ap)     = make_uint4(bf2pack(h0.x, h0.y), bf2pack(h0.z, h0.w),
                                       bf2pack(h1.x, h1.y), bf2pack(h1.z, h1.w));
        *(uint4*)(ap + 4) = make_uint4(bf2pack(h2.x, h2.y), bf2pack(h2.z, h2.w),
                                       bf2pack(h3.x, h3.y), bf2pack(h3.z, h3.w));
        const float* bp = Bw + ((size_t)aid * DOUT + (n0 + lrow)) * 16;
        float4 w0 = *(const float4*)(bp);
        float4 w1 = *(const float4*)(bp + 4);
        float4 w2 = *(const float4*)(bp + 8);
        float4 w3 = *(const float4*)(bp + 12);
        unsigned* bq = (unsigned*)&b_sm[0][lrow * LDA];
        *(uint4*)(bq)     = make_uint4(bf2pack(w0.x, w0.y), bf2pack(w0.z, w0.w),
                                       bf2pack(w1.x, w1.y), bf2pack(w1.z, w1.w));
        *(uint4*)(bq + 4) = make_uint4(bf2pack(w2.x, w2.y), bf2pack(w2.z, w2.w),
                                       bf2pack(w3.x, w3.y), bf2pack(w3.z, w3.w));
    } else {
        uint4 z = make_uint4(0u, 0u, 0u, 0u);
        unsigned* ap = (unsigned*)&a_sm[0][lrow * LDA + 16];
        *(uint4*)(ap) = z; *(uint4*)(ap + 4) = z;
        unsigned* bq = (unsigned*)&b_sm[0][lrow * LDA + 16];
        *(uint4*)(bq) = z; *(uint4*)(bq + 4) = z;
    }
    __syncthreads();

    {
        Frag fa[2], fb[4];
        #pragma unroll
        for (int i = 0; i < 2; ++i) {
            const unsigned short* base = &a_sm[0][(wm * 32 + i * 16 + tm) * LDA];
            *(uint4*)&fa[i].u[0] = *(const uint4*)(base + sel * 8);
            *(uint4*)&fa[i].u[4] = *(const uint4*)(base + 16 + sel * 8);
        }
        #pragma unroll
        for (int j = 0; j < 4; ++j) {
            const unsigned short* base = &b_sm[0][(wn * 64 + j * 16 + tm) * LDA];
            *(uint4*)&fb[j].u[0] = *(const uint4*)(base + sel * 8);
            *(uint4*)&fb[j].u[4] = *(const uint4*)(base + 16 + sel * 8);
        }
        #pragma unroll
        for (int i = 0; i < 2; ++i)
            #pragma unroll
            for (int j = 0; j < 4; ++j)
                acc[i][j] = __builtin_amdgcn_wmma_f32_16x16x32_bf16(
                    false, fa[i].v, false, fb[j].v, (short)0, acc[i][j],
                    false, false);
    }

    float bj[4];
    #pragma unroll
    for (int j = 0; j < 4; ++j)
        bj[j] = bias[n0 + wn * 64 + j * 16 + tm];

    #pragma unroll
    for (int i = 0; i < 2; ++i) {
        const int rbase = m0 + wm * 32 + i * 16 + sel * 8;
        #pragma unroll
        for (int j = 0; j < 4; ++j) {
            const int col = n0 + wn * 64 + j * 16 + tm;
            #pragma unroll
            for (int v = 0; v < 8; ++v)
                out[(size_t)(rbase + v) * DOUT + col] = acc[i][j][v] + bj[j];
        }
    }
}

// ---------------------------------------------------------------------------
// Kernel B (fallback): f32 operands, convert in-kernel, WMMA compute
// ---------------------------------------------------------------------------
__global__ __launch_bounds__(256) void lora_gemm_kernel(
    const float* __restrict__ x,
    const float* __restrict__ W,
    const float* __restrict__ bias,
    const float* __restrict__ Bw,
    const int*   __restrict__ adapter_id,
    const float* __restrict__ h,
    float*       __restrict__ out)
{
    __shared__ unsigned short a_sm[BM * LDA];
    __shared__ unsigned short b_sm[BN * LDA];

    const int t    = threadIdx.x;
    const int lane = t & 31;
    const int wave = t >> 5;
    const int tm   = lane & 15;
    const int sel  = lane >> 4;
    const int wm   = wave & 3;
    const int wn   = wave >> 2;

    const int m0 = blockIdx.y * BM;
    const int n0 = blockIdx.x * BN;

    const int lrow  = t >> 1;
    const int lhalf = t & 1;

    const float* xg = x + (size_t)(m0 + lrow) * DIN + lhalf * 16;
    const float* wg = W + (size_t)(n0 + lrow) * DIN + lhalf * 16;
    unsigned* a_st = (unsigned*)&a_sm[lrow * LDA + lhalf * 16];
    unsigned* b_st = (unsigned*)&b_sm[lrow * LDA + lhalf * 16];

    v8f zf = {};
    v8f acc[2][4];
    #pragma unroll
    for (int i = 0; i < 2; ++i)
        #pragma unroll
        for (int j = 0; j < 4; ++j) acc[i][j] = zf;

    for (int k0 = 0; k0 < DIN; k0 += BK) {
        float4 r0 = *(const float4*)(xg + k0);
        float4 r1 = *(const float4*)(xg + k0 + 4);
        float4 r2 = *(const float4*)(xg + k0 + 8);
        float4 r3 = *(const float4*)(xg + k0 + 12);
        float4 s0 = *(const float4*)(wg + k0);
        float4 s1 = *(const float4*)(wg + k0 + 4);
        float4 s2 = *(const float4*)(wg + k0 + 8);
        float4 s3 = *(const float4*)(wg + k0 + 12);
        if (k0 + BK < DIN) {
            __builtin_prefetch(xg + k0 + BK, 0, 3);
            __builtin_prefetch(wg + k0 + BK, 0, 3);
        }
        uint4 pa0 = make_uint4(bf2pack(r0.x, r0.y), bf2pack(r0.z, r0.w),
                               bf2pack(r1.x, r1.y), bf2pack(r1.z, r1.w));
        uint4 pa1 = make_uint4(bf2pack(r2.x, r2.y), bf2pack(r2.z, r2.w),
                               bf2pack(r3.x, r3.y), bf2pack(r3.z, r3.w));
        uint4 pb0 = make_uint4(bf2pack(s0.x, s0.y), bf2pack(s0.z, s0.w),
                               bf2pack(s1.x, s1.y), bf2pack(s1.z, s1.w));
        uint4 pb1 = make_uint4(bf2pack(s2.x, s2.y), bf2pack(s2.z, s2.w),
                               bf2pack(s3.x, s3.y), bf2pack(s3.z, s3.w));

        __syncthreads();
        *(uint4*)(a_st)     = pa0;
        *(uint4*)(a_st + 4) = pa1;
        *(uint4*)(b_st)     = pb0;
        *(uint4*)(b_st + 4) = pb1;
        __syncthreads();

        Frag fa[2], fb[4];
        #pragma unroll
        for (int i = 0; i < 2; ++i) {
            const unsigned short* base = &a_sm[(wm * 32 + i * 16 + tm) * LDA];
            *(uint4*)&fa[i].u[0] = *(const uint4*)(base + sel * 8);
            *(uint4*)&fa[i].u[4] = *(const uint4*)(base + 16 + sel * 8);
        }
        #pragma unroll
        for (int j = 0; j < 4; ++j) {
            const unsigned short* base = &b_sm[(wn * 64 + j * 16 + tm) * LDA];
            *(uint4*)&fb[j].u[0] = *(const uint4*)(base + sel * 8);
            *(uint4*)&fb[j].u[4] = *(const uint4*)(base + 16 + sel * 8);
        }
        #pragma unroll
        for (int i = 0; i < 2; ++i)
            #pragma unroll
            for (int j = 0; j < 4; ++j)
                acc[i][j] = __builtin_amdgcn_wmma_f32_16x16x32_bf16(
                    false, fa[i].v, false, fb[j].v, (short)0, acc[i][j],
                    false, false);
    }

    const int aid = adapter_id[m0 >> 11];
    __syncthreads();
    if (lhalf == 0) {
        const float* hp = h + (size_t)(m0 + lrow) * 16;
        float4 h0 = *(const float4*)(hp);
        float4 h1 = *(const float4*)(hp + 4);
        float4 h2 = *(const float4*)(hp + 8);
        float4 h3 = *(const float4*)(hp + 12);
        unsigned* ap = (unsigned*)&a_sm[lrow * LDA];
        *(uint4*)(ap)     = make_uint4(bf2pack(h0.x, h0.y), bf2pack(h0.z, h0.w),
                                       bf2pack(h1.x, h1.y), bf2pack(h1.z, h1.w));
        *(uint4*)(ap + 4) = make_uint4(bf2pack(h2.x, h2.y), bf2pack(h2.z, h2.w),
                                       bf2pack(h3.x, h3.y), bf2pack(h3.z, h3.w));
        const float* bp = Bw + ((size_t)aid * DOUT + (n0 + lrow)) * 16;
        float4 w0 = *(const float4*)(bp);
        float4 w1 = *(const float4*)(bp + 4);
        float4 w2 = *(const float4*)(bp + 8);
        float4 w3 = *(const float4*)(bp + 12);
        unsigned* bq = (unsigned*)&b_sm[lrow * LDA];
        *(uint4*)(bq)     = make_uint4(bf2pack(w0.x, w0.y), bf2pack(w0.z, w0.w),
                                       bf2pack(w1.x, w1.y), bf2pack(w1.z, w1.w));
        *(uint4*)(bq + 4) = make_uint4(bf2pack(w2.x, w2.y), bf2pack(w2.z, w2.w),
                                       bf2pack(w3.x, w3.y), bf2pack(w3.z, w3.w));
    } else {
        uint4 z = make_uint4(0u, 0u, 0u, 0u);
        unsigned* ap = (unsigned*)&a_sm[lrow * LDA + 16];
        *(uint4*)(ap) = z; *(uint4*)(ap + 4) = z;
        unsigned* bq = (unsigned*)&b_sm[lrow * LDA + 16];
        *(uint4*)(bq) = z; *(uint4*)(bq + 4) = z;
    }
    __syncthreads();

    {
        Frag fa[2], fb[4];
        #pragma unroll
        for (int i = 0; i < 2; ++i) {
            const unsigned short* base = &a_sm[(wm * 32 + i * 16 + tm) * LDA];
            *(uint4*)&fa[i].u[0] = *(const uint4*)(base + sel * 8);
            *(uint4*)&fa[i].u[4] = *(const uint4*)(base + 16 + sel * 8);
        }
        #pragma unroll
        for (int j = 0; j < 4; ++j) {
            const unsigned short* base = &b_sm[(wn * 64 + j * 16 + tm) * LDA];
            *(uint4*)&fb[j].u[0] = *(const uint4*)(base + sel * 8);
            *(uint4*)&fb[j].u[4] = *(const uint4*)(base + 16 + sel * 8);
        }
        #pragma unroll
        for (int i = 0; i < 2; ++i)
            #pragma unroll
            for (int j = 0; j < 4; ++j)
                acc[i][j] = __builtin_amdgcn_wmma_f32_16x16x32_bf16(
                    false, fa[i].v, false, fb[j].v, (short)0, acc[i][j],
                    false, false);
    }

    float bj[4];
    #pragma unroll
    for (int j = 0; j < 4; ++j)
        bj[j] = bias[n0 + wn * 64 + j * 16 + tm];

    #pragma unroll
    for (int i = 0; i < 2; ++i) {
        const int rbase = m0 + wm * 32 + i * 16 + sel * 8;
        #pragma unroll
        for (int j = 0; j < 4; ++j) {
            const int col = n0 + wn * 64 + j * 16 + tm;
            #pragma unroll
            for (int v = 0; v < 8; ++v)
                out[(size_t)(rbase + v) * DOUT + col] = acc[i][j][v] + bj[j];
        }
    }
}

// ---------------------------------------------------------------------------
extern "C" void kernel_launch(void* const* d_in, const int* in_sizes, int n_in,
                              void* d_out, int out_size, void* d_ws, size_t ws_size,
                              hipStream_t stream) {
    (void)in_sizes; (void)n_in; (void)out_size;
    const float* x          = (const float*)d_in[0];
    const float* W          = (const float*)d_in[1];
    const float* bias       = (const float*)d_in[2];
    const float* Alora      = (const float*)d_in[3];
    const float* Bw         = (const float*)d_in[4];
    const float* drop_u     = (const float*)d_in[5];
    const int*   adapter_id = (const int*)d_in[6];
    float* out = (float*)d_out;
    float* h   = (float*)d_ws;                       // [16384,16] f32 = 1 MB

    const size_t xb_off = 2ull << 20;                // h region padded to 2 MB
    const size_t xb_bytes = (size_t)MROWS * DIN * 2; // 128 MB
    const size_t wb_bytes = (size_t)DOUT * DIN * 2;  // 32 MB
    const size_t need = xb_off + xb_bytes + wb_bytes;

    lora_h_kernel<<<dim3(MROWS / 16 / 8), dim3(256), 0, stream>>>(
        x, Alora, drop_u, adapter_id, h);

    if (ws_size >= need) {
        unsigned short* xb = (unsigned short*)((char*)d_ws + xb_off);
        unsigned short* wb = xb + (size_t)MROWS * DIN;
        // one-time f32 -> bf16 of the GEMM operands
        cvt_bf16_kernel<<<dim3((MROWS * (DIN / 8)) / 256), dim3(256), 0, stream>>>(
            x, xb, MROWS * (DIN / 8));
        cvt_bf16_kernel<<<dim3((DOUT * (DIN / 8)) / 256), dim3(256), 0, stream>>>(
            W, wb, DOUT * (DIN / 8));
        lora_gemm_tdm_kernel<<<dim3(DOUT / BN, MROWS / BM), dim3(256), 0, stream>>>(
            xb, wb, bias, Bw, adapter_id, h, out);
    } else {
        lora_gemm_kernel<<<dim3(DOUT / BN, MROWS / BM), dim3(256), 0, stream>>>(
            x, W, bias, Bw, adapter_id, h, out);
    }
}
